// VFNet_60756607369853
// MI455X (gfx1250) — matrix operand order repaired and checked by reading, not compiled, not run on hardware
//
#include <hip/hip_runtime.h>

// ---------------------------------------------------------------------------
// VFNet voxel-projection + masked-MLP, fused for gfx1250 (MI455X, wave32).
// GEMM path: v_wmma_f32_16x16x32_bf16, M=128 (64 "count==1" rows + 64
// "count==2" rows), K=130 padded to 160 (5 chunks of 32), N=200000.
// Round 3: software-pipeline the A-fragment LDS loads (double-buffered per
// K-chunk) so ds latency is not serialized against every WMMA.
// ---------------------------------------------------------------------------

typedef __bf16 bf16_t;
typedef __bf16 v8bf  __attribute__((ext_vector_type(8)));
typedef __bf16 v16bf __attribute__((ext_vector_type(16)));
typedef float  v8f   __attribute__((ext_vector_type(8)));

#define NCAMS   6
#define CF      64
#define IH      44
#define IW      80
#define NVOX    200000
#define KTOT    130          // 65 (cams 0-2) + 65 (cams 3-5)
#define KSTR    168          // per-voxel K stride in LDS (bf16), 16B aligned
#define TILE_N  128          // voxels per workgroup
#define NPACKA  (8*5*32*16)  // 8 M-tiles x 5 K-chunks x 32 lanes x 16 bf16

// ---------------------------------------------------------------------------
// Kernel 1: pack combined weights [128 x 130] into the exact CDNA5 16-bit
// A-matrix fragment layout (ISA 7.12.2): lane L holds M = mtile*16 + L%16;
// VGPR v (halves h) holds K = (v>=4?16:0) + (L>=16?8:0) + 2*(v&3) + h.
// Rows 0..63  = W_no applied to both K halves (y_no = W_no*f1 + W_no*f2).
// Rows 64..127= W_o  (columns 0..129 directly).
// ---------------------------------------------------------------------------
__global__ void vf_pack_weights(const float* __restrict__ W_no,
                                const float* __restrict__ W_o,
                                unsigned short* __restrict__ packedA) {
  int idx = blockIdx.x * 256 + threadIdx.x;
  if (idx >= NPACKA) return;
  int j    = idx & 15;
  int lane = (idx >> 4) & 31;
  int mk   = idx >> 9;           // mtile*5 + kc
  int kc   = mk % 5;
  int mt   = mk / 5;
  int v = j >> 1, h = j & 1;
  int kl = ((v >= 4) ? 16 : 0) + ((lane >= 16) ? 8 : 0) + 2 * (v & 3) + h;
  int gk = kc * 32 + kl;
  int M  = mt * 16 + (lane & 15);
  float wv = 0.f;
  if (gk < KTOT) {
    if (M < 64) wv = W_no[M * 65 + (gk < 65 ? gk : gk - 65)];
    else        wv = W_o[(M - 64) * KTOT + gk];
  }
  bf16_t b = (bf16_t)wv;
  packedA[idx] = __builtin_bit_cast(unsigned short, b);
}

// ---------------------------------------------------------------------------
// Kernel 2: feats (6,64,44,80) -> channel-last (6,44,80,64) so every bilinear
// corner gather is 64 contiguous floats (16x float4, 256B -> cache friendly).
// ---------------------------------------------------------------------------
__global__ void vf_transpose_feats(const float* __restrict__ feats,
                                   float* __restrict__ featsT) {
  int idx = blockIdx.x * 256 + threadIdx.x;
  if (idx >= NCAMS * CF * IH * IW) return;
  int c = idx & 63;
  int t = idx >> 6;              // (cam*IH + y)*IW + x
  int x = t % IW;
  int y = (t / IW) % IH;
  int cam = t / (IW * IH);
  featsT[idx] = feats[((cam * CF + c) * IH + y) * IW + x];
}

// ---------------------------------------------------------------------------
// Kernel 3: fused projection + bilinear gather + WMMA MLP + masked epilogue.
// 256 threads = 8 wave32. Stage 1: thread t -> voxel t%128, cam-half t/128.
// Stage 2: wave w owns voxel column [w*16, w*16+16), M loop over 4 tile pairs.
// ---------------------------------------------------------------------------
__global__ __launch_bounds__(256)
void vf_fused(const float* __restrict__ featsT,       // (6,44,80,64)
              const float* __restrict__ maskimg,      // (6,1,44,80)
              const float* __restrict__ intr,         // (6,4,4)
              const float* __restrict__ extr,         // (6,4,4)
              const unsigned short* __restrict__ packedA,
              const float* __restrict__ b_no,
              const float* __restrict__ b_o,
              float* __restrict__ out) {
  __shared__ unsigned short tile[TILE_N * KSTR];          // bf16 feature tile [N][K]
  __shared__ __align__(32) unsigned short aPack[NPACKA];  // staged A fragments (40 KB)
  __shared__ int cnt2[2][TILE_N];

  const int tid   = threadIdx.x;
  const int nbase = blockIdx.x * TILE_N;

  // Zero the feature tile (covers K padding 130..159 and tail voxels) and
  // stage the packed-A weight fragments global -> LDS (coalesced uint4 copy).
  {
    unsigned int* tz = (unsigned int*)tile;
    for (int i = tid; i < TILE_N * KSTR / 2; i += 256) tz[i] = 0u;
    const uint4* src = (const uint4*)packedA;
    uint4* dst = (uint4*)aPack;
#pragma unroll
    for (int i = 0; i < NPACKA / 8 / 256; ++i)    // 2560 uint4 / 256 thr = 10
      dst[tid + i * 256] = src[tid + i * 256];
  }
  __syncthreads();

  // ---------------- Stage 1: project + gather + accumulate -----------------
  {
    const int v    = tid & (TILE_N - 1);
    const int half = tid >> 7;              // 0: cams 0-2 -> K 0..64 ; 1: cams 3-5 -> K 65..129
    const int n    = nbase + v;
    float acc[CF];
#pragma unroll
    for (int c = 0; c < CF; ++c) acc[c] = 0.f;
    float accz = 0.f;
    int count = 0;

    if (n < NVOX) {
      const int xi = n % 100, yi = (n / 100) % 100, zi = n / 10000;
      const float xw = -50.f + 1.0f * (float)xi;
      const float yw = -50.f + 1.0f * (float)yi;
      const float zw = -15.f + 1.5f * (float)zi;

      for (int cam = half * 3; cam < half * 3 + 3; ++cam) {
        const float* E  = extr + cam * 16;
        const float* Km = intr + cam * 16;
        const float vx = E[0] * xw + E[1] * yw + E[2]  * zw + E[3];
        const float vy = E[4] * xw + E[5] * yw + E[6]  * zw + E[7];
        const float vz = E[8] * xw + E[9] * yw + E[10] * zw + E[11];
        const float cx = Km[0] * vx + Km[1] * vy + Km[2]  * vz;
        const float cy = Km[4] * vx + Km[5] * vy + Km[6]  * vz;
        const float cz = Km[8] * vx + Km[9] * vy + Km[10] * vz;
        const float pz = cz + 1e-8f;
        const float px = cx / pz;
        const float py = cy / pz;
        const float gx = (px * (1.f / (IW - 1)) - 0.5f) * 2.f;
        const float gy = (py * (1.f / (IH - 1)) - 0.5f) * 2.f;

        // nearest-sampled mask (RNE round, in-range validity)
        const int xn = (int)rintf(px);
        const int yn = (int)rintf(py);
        const bool minb = (xn >= 0) && (xn < IW) && (yn >= 0) && (yn < IH);
        float ms = 0.f;
        if (minb) ms = maskimg[cam * IH * IW + yn * IW + xn];
        const bool valid = (ms > 0.5f) && (vz > 0.f) &&
                           !(fabsf(gx) > 1.f || fabsf(gy) > 1.f);
        if (valid) {
          ++count;
          const float x0f = floorf(px), y0f = floorf(py);
          const int x0 = (int)x0f, y0 = (int)y0f, x1 = x0 + 1, y1 = y0 + 1;
          const float wx1 = px - x0f, wx0 = 1.f - wx1;
          const float wy1 = py - y0f, wy0 = 1.f - wy1;
          const bool v00 = (x0 >= 0) && (y0 >= 0);            // x0<IW,y0<IH implied by oob mask
          const bool v10 = (x1 < IW) && (y0 >= 0);
          const bool v01 = (x0 >= 0) && (y1 < IH);
          const bool v11 = (x1 < IW) && (y1 < IH);
          const float* fb = featsT + (size_t)cam * (IH * IW * CF);
          const float* p00 = v00 ? fb + (size_t)(y0 * IW + x0) * CF : fb;
          const float* p10 = v10 ? fb + (size_t)(y0 * IW + x1) * CF : fb;
          const float* p01 = v01 ? fb + (size_t)(y1 * IW + x0) * CF : fb;
          const float* p11 = v11 ? fb + (size_t)(y1 * IW + x1) * CF : fb;
          const float w00 = v00 ? wx0 * wy0 : 0.f;
          const float w10 = v10 ? wx1 * wy0 : 0.f;
          const float w01 = v01 ? wx0 * wy1 : 0.f;
          const float w11 = v11 ? wx1 * wy1 : 0.f;
#pragma unroll
          for (int cc = 0; cc < 16; ++cc) {
            const float4 a0 = ((const float4*)p00)[cc];
            const float4 a1 = ((const float4*)p10)[cc];
            const float4 a2 = ((const float4*)p01)[cc];
            const float4 a3 = ((const float4*)p11)[cc];
            acc[4 * cc + 0] += w00 * a0.x + w10 * a1.x + w01 * a2.x + w11 * a3.x;
            acc[4 * cc + 1] += w00 * a0.y + w10 * a1.y + w01 * a2.y + w11 * a3.y;
            acc[4 * cc + 2] += w00 * a0.z + w10 * a1.z + w01 * a2.z + w11 * a3.z;
            acc[4 * cc + 3] += w00 * a0.w + w10 * a1.w + w01 * a2.w + w11 * a3.w;
          }
          accz += vz * (1.f / 100.f);   // z-channel (vloc_z / VOX_SIZE[0]) * valid
        }
      }
    }
    unsigned short* dst = tile + v * KSTR + half * 65;
#pragma unroll
    for (int c = 0; c < CF; ++c) {
      bf16_t b = (bf16_t)acc[c];
      dst[c] = __builtin_bit_cast(unsigned short, b);
    }
    {
      bf16_t b = (bf16_t)accz;
      dst[CF] = __builtin_bit_cast(unsigned short, b);
    }
    cnt2[half][v] = count;
  }
  __syncthreads();

  // ---------------- Stage 2: WMMA GEMM + fused epilogue ---------------------
  {
    const int wave = tid >> 5;          // n-tile id (0..7)
    const int lane = tid & 31;
    const int ncol = (wave << 4) + (lane & 15);
    const int klo  = (lane & 16) >> 1;  // B K-half: lanes 0-15 -> K+0/+16, 16-31 -> K+8/+24

    // Cache all 5 B fragments (K chunks of 32) for this wave's voxel column.
    v16bf bfrag[5];
#pragma unroll
    for (int kc = 0; kc < 5; ++kc) {
      const unsigned short* bp = tile + ncol * KSTR + kc * 32 + klo;
      v8bf lo = *(const v8bf*)bp;
      v8bf hi = *(const v8bf*)(bp + 16);
      bfrag[kc] = __builtin_shufflevector(lo, hi, 0, 1, 2, 3, 4, 5, 6, 7,
                                                  8, 9, 10, 11, 12, 13, 14, 15);
    }

    const int cnt  = cnt2[0][ncol] + cnt2[1][ncol];
    const float fno = (cnt == 1) ? 1.f : 0.f;
    const float fo  = (cnt == 2) ? 1.f : 0.f;
    const int nglob  = nbase + ncol;
    const int rowoff = (lane & 16) ? 8 : 0;   // C/D layout: VGPR r -> M=r (+8 hi lanes)

    // A-fragment LDS load (fragment id f = mtile*5 + kc; 32B per lane).
    auto ldA = [&](int f) -> v16bf {
      return *(const v16bf*)(aPack + (f * 32 + lane) * 16);
    };

    for (int m = 0; m < 4; ++m) {             // M tile pair: rows m*16.. (no) / 64+m*16.. (o)
      v8f acc_no = {};
      v8f acc_o  = {};
      // Software pipeline: keep next chunk's A fragments in flight while the
      // current pair of WMMAs issues (distinct SSA values -> no ds serialization).
      v16bf a_no = ldA(m * 5);
      v16bf a_o  = ldA((m + 4) * 5);
#pragma unroll
      for (int kc = 0; kc < 5; ++kc) {
        v16bf a_no_nxt = a_no;
        v16bf a_o_nxt  = a_o;
        if (kc < 4) {
          a_no_nxt = ldA(m * 5 + kc + 1);
          a_o_nxt  = ldA((m + 4) * 5 + kc + 1);
        }
        acc_no = __builtin_amdgcn_wmma_f32_16x16x32_bf16(
            false, a_no, false, bfrag[kc], (short)0, acc_no, false, false);
        acc_o = __builtin_amdgcn_wmma_f32_16x16x32_bf16(
            false, a_o, false, bfrag[kc], (short)0, acc_o, false, false);
        a_no = a_no_nxt;
        a_o  = a_o_nxt;
      }
      if (nglob < NVOX) {
#pragma unroll
        for (int r = 0; r < 8; ++r) {
          const int c = m * 16 + r + rowoff;
          float v1 = acc_no[r] + b_no[c];
          float v2 = acc_o[r]  + b_o[c];
          v1 = v1 > 0.f ? v1 : 0.f;
          v2 = v2 > 0.f ? v2 : 0.f;
          out[(size_t)c * NVOX + nglob] = v1 * fno + v2 * fo;
        }
      }
    }
  }
}

// ---------------------------------------------------------------------------
extern "C" void kernel_launch(void* const* d_in, const int* in_sizes, int n_in,
                              void* d_out, int out_size, void* d_ws, size_t ws_size,
                              hipStream_t stream) {
  (void)in_sizes; (void)n_in; (void)out_size; (void)ws_size;
  const float* feats = (const float*)d_in[0];   // (1,6,64,44,80)
  const float* mask  = (const float*)d_in[1];   // (1,6,1,44,80)
  const float* intr  = (const float*)d_in[2];   // (1,6,4,4)
  const float* extr  = (const float*)d_in[3];   // (1,6,4,4)
  const float* W_no  = (const float*)d_in[4];   // (64,65)
  const float* b_no  = (const float*)d_in[5];   // (64)
  const float* W_o   = (const float*)d_in[6];   // (64,130)
  const float* b_o   = (const float*)d_in[7];   // (64)
  float* out = (float*)d_out;                   // (1,64,200000)

  unsigned short* packedA = (unsigned short*)d_ws;                 // 40 KB
  float* featsT = (float*)((char*)d_ws + 65536);                   // 5.4 MB

  vf_pack_weights<<<(NPACKA + 255) / 256, 256, 0, stream>>>(W_no, W_o, packedA);
  vf_transpose_feats<<<(NCAMS * CF * IH * IW + 255) / 256, 256, 0, stream>>>(feats, featsT);
  vf_fused<<<(NVOX + TILE_N - 1) / TILE_N, 256, 0, stream>>>(
      featsT, mask, intr, extr, packedA, b_no, b_o, out);
}
